// SelfAttentionHead_49314814492954
// MI455X (gfx1250) — matrix operand order repaired and checked by reading, not compiled, not run on hardware
//
#include <hip/hip_runtime.h>

typedef __attribute__((ext_vector_type(16))) __bf16        v16bf;
typedef __attribute__((ext_vector_type(8)))  float         v8f;
typedef __attribute__((ext_vector_type(4)))  float         v4f;
typedef __attribute__((ext_vector_type(4)))  unsigned int  v4u;

#define HEAD   64
#define EMBED  1024
#define SEQ    4096
#define BATCH  4
#define NTOK   (BATCH * SEQ)                       // 16384 rows
#define PROJ_ELEMS ((size_t)NTOK * HEAD)           // bf16 elems per q/k/v buffer
#define WT_OFF (3 * PROJ_ELEMS)                    // transposed weights offset

static __device__ __forceinline__ v8f wmma_bf16(v16bf a, v16bf b, v8f c) {
  // D = A(16x32 bf16) * B(32x16 bf16) + C(16x16 f32)
  return __builtin_amdgcn_wmma_f32_16x16x32_bf16(
      false, a, false, b, (short)0, c, false, false);
}

// two 8-bf16 runs at p and p+16 (16B-aligned) -> one B/A fragment half
static __device__ __forceinline__ v16bf load_bf16x16(const __bf16* p) {
  union { v4u q[2]; v16bf v; } u;
  u.q[0] = *reinterpret_cast<const v4u*>(p);
  u.q[1] = *reinterpret_cast<const v4u*>(p + 16);
  return u.v;
}

// ---------------------------------------------------------------------------
// Kernel 0: transpose W[k][n] f32 -> Wt[w][n][k] bf16 so B-fragment loads are
// contiguous along K.  3 * 64 * 1024 elements.
// ---------------------------------------------------------------------------
__global__ __launch_bounds__(256) void prep_w_kernel(
    const float* __restrict__ Wk, const float* __restrict__ Wq,
    const float* __restrict__ Wv, unsigned short* __restrict__ ws)
{
  __bf16* Wt = reinterpret_cast<__bf16*>(ws) + WT_OFF;
  const int idx = blockIdx.x * 256 + threadIdx.x;   // 0 .. 3*64*1024-1
  const int w   = idx >> 16;                        // HEAD*EMBED = 65536
  const int rem = idx & 65535;
  const int n   = rem >> 10;
  const int k   = rem & 1023;
  const float* W = (w == 0) ? Wk : (w == 1) ? Wq : Wv;
  Wt[idx] = (__bf16)W[k * HEAD + n];
}

// ---------------------------------------------------------------------------
// Kernel 1: q/k/v projections.  x[16384,1024] f32 @ Wt[64,1024] bf16.
// One wave = one 16x64 output tile; blockIdx.y selects {k,q,v}.
// q pre-scaled by H^-0.5 = 0.125.  v stored transposed: vT[b][h][s].
// ---------------------------------------------------------------------------
__global__ __launch_bounds__(128) void qkv_proj_kernel(
    const float* __restrict__ x, unsigned short* __restrict__ ws)
{
  __bf16* wsbf = reinterpret_cast<__bf16*>(ws);
  __bf16* outk = wsbf;
  __bf16* outq = wsbf + PROJ_ELEMS;
  __bf16* outv = wsbf + 2 * PROJ_ELEMS;            // transposed layout
  const __bf16* Wt = wsbf + WT_OFF;

  const int wave = threadIdx.x >> 5;
  const int lane = threadIdx.x & 31;
  const int half = lane >> 4;
  const int l16  = lane & 15;
  const int tile = blockIdx.x * 4 + wave;          // 0..1023
  const int which = blockIdx.y;                    // 0=k, 1=q, 2=v

  const __bf16* WtW = Wt + (size_t)which * HEAD * EMBED;
  const float scale = (which == 1) ? 0.125f : 1.0f;
  const size_t row = (size_t)tile * 16 + l16;      // A-matrix row = lane%16

  v8f acc[4] = {};

  for (int k0 = 0; k0 < EMBED; k0 += 32) {
    // A fragment (16x32 bf16): two contiguous 8-float runs per lane
    const float* xp = x + row * EMBED + k0;
    v4f c0 = *reinterpret_cast<const v4f*>(xp + 8 * half);
    v4f c1 = *reinterpret_cast<const v4f*>(xp + 8 * half + 4);
    v4f c2 = *reinterpret_cast<const v4f*>(xp + 16 + 8 * half);
    v4f c3 = *reinterpret_cast<const v4f*>(xp + 16 + 8 * half + 4);
    v16bf a;
#pragma unroll
    for (int i = 0; i < 4; ++i) {
      a[i]      = (__bf16)c0[i];
      a[4 + i]  = (__bf16)c1[i];
      a[8 + i]  = (__bf16)c2[i];
      a[12 + i] = (__bf16)c3[i];
    }
#pragma unroll
    for (int t = 0; t < 4; ++t) {
      // B fragment (32x16): element e -> K = 8*half + e (e<8), 16 + 8*half + (e-8) (e>=8)
      v16bf b = load_bf16x16(WtW + (size_t)(t * 16 + l16) * EMBED + k0 + 8 * half);
      acc[t] = wmma_bf16(a, b, acc[t]);
    }
  }

  // C/D layout: VGPR v holds row = v + 8*half, col = l16 (per N-tile)
  if (which == 2) {
#pragma unroll
    for (int t = 0; t < 4; ++t)
#pragma unroll
      for (int v = 0; v < 8; ++v) {
        const int r  = tile * 16 + v + 8 * half;   // global token 0..16383
        const int bb = r >> 12;                    // batch
        const int s  = r & (SEQ - 1);
        outv[(size_t)bb * HEAD * SEQ + (size_t)(t * 16 + l16) * SEQ + s] =
            (__bf16)acc[t][v];
      }
  } else {
    __bf16* out = (which == 0) ? outk : outq;
#pragma unroll
    for (int t = 0; t < 4; ++t)
#pragma unroll
      for (int v = 0; v < 8; ++v) {
        const size_t r = (size_t)tile * 16 + v + 8 * half;
        out[r * HEAD + t * 16 + l16] = (__bf16)(acc[t][v] * scale);
      }
  }
}

// ---------------------------------------------------------------------------
// Kernel 2: causal flash attention, 32 keys per iteration (dense PV WMMA).
// One wave owns one (batch, 16-query-row) tile.
// ---------------------------------------------------------------------------
__global__ __launch_bounds__(256) void flash_attn_kernel(
    const unsigned short* __restrict__ ws, float* __restrict__ z)
{
  const __bf16* wsbf = reinterpret_cast<const __bf16*>(ws);
  const __bf16* kbuf = wsbf;
  const __bf16* qbuf = wsbf + PROJ_ELEMS;
  const __bf16* vT   = wsbf + 2 * PROJ_ELEMS;      // [b][h][s]

  __shared__ float lds[8][16][32];                 // per-wave P scratch (16 KB)

  const int wave = threadIdx.x >> 5;
  const int lane = threadIdx.x & 31;
  const int half = lane >> 4;
  const int l16  = lane & 15;

  const int gtile = blockIdx.x * 8 + wave;         // 0..1023
  const int b  = gtile >> 8;
  const int qt = gtile & 255;
  const size_t base  = (size_t)b * SEQ * HEAD;
  const size_t vbase = (size_t)b * HEAD * SEQ;
  const int    qrow0 = qt * 16;

  // Q A-fragments (h 0..31 and 32..63); q pre-scaled by 0.125
  const __bf16* qp = qbuf + base + (size_t)(qrow0 + l16) * HEAD;
  v16bf qa0, qa1;
  {
    union { v4u q[2]; v16bf v; } u0, u1;
    u0.q[0] = *reinterpret_cast<const v4u*>(qp + 8 * half);
    u0.q[1] = *reinterpret_cast<const v4u*>(qp + 16 + 8 * half);
    u1.q[0] = *reinterpret_cast<const v4u*>(qp + 32 + 8 * half);
    u1.q[1] = *reinterpret_cast<const v4u*>(qp + 48 + 8 * half);
    qa0 = u0.v;  qa1 = u1.v;
  }

  float m[8], l[8];
#pragma unroll
  for (int v = 0; v < 8; ++v) { m[v] = -INFINITY; l[v] = 0.0f; }
  v8f acc[4] = {};

  const int nchunk = (qt >> 1) + 1;                // ceil((qt+1)*16 / 32)
  for (int kc = 0; kc < nchunk; ++kc) {
    const int j0 = kc * 32;
    const int j1 = j0 + 16;

    // ---- S = Q K^T for two 16-key subtiles (contraction over h=64) ----
    const __bf16* kp0 = kbuf + base + (size_t)(j0 + l16) * HEAD;
    const __bf16* kp1 = kbuf + base + (size_t)(j1 + l16) * HEAD;
    v16bf kb00 = load_bf16x16(kp0 + 8 * half);       // subtile0, h 0..31
    v16bf kb01 = load_bf16x16(kp0 + 32 + 8 * half);  // subtile0, h 32..63
    v16bf kb10 = load_bf16x16(kp1 + 8 * half);
    v16bf kb11 = load_bf16x16(kp1 + 32 + 8 * half);

    v8f s0 = {}, s1 = {};
    s0 = wmma_bf16(qa0, kb00, s0);
    s0 = wmma_bf16(qa1, kb01, s0);
    s1 = wmma_bf16(qa0, kb10, s1);
    s1 = wmma_bf16(qa1, kb11, s1);

    // ---- causal mask (uniform branch; only last chunk needs it) ----
    if (j0 + 15 > qrow0) {
#pragma unroll
      for (int v = 0; v < 8; ++v)
        if (j0 + l16 > qrow0 + v + 8 * half) s0[v] = -INFINITY;
    }
    if (j1 + 15 > qrow0) {
#pragma unroll
      for (int v = 0; v < 8; ++v)
        if (j1 + l16 > qrow0 + v + 8 * half) s1[v] = -INFINITY;
    }

    // ---- online softmax over 32 columns ----
    float alpha[8];
#pragma unroll
    for (int v = 0; v < 8; ++v) {
      float mx = fmaxf(s0[v], s1[v]);
#pragma unroll
      for (int off = 8; off >= 1; off >>= 1)
        mx = fmaxf(mx, __shfl_xor(mx, off, 32));
      const float mnew = fmaxf(m[v], mx);
      alpha[v] = __expf(m[v] - mnew);
      const float p0 = __expf(s0[v] - mnew);
      const float p1 = __expf(s1[v] - mnew);
      float ps = p0 + p1;
#pragma unroll
      for (int off = 8; off >= 1; off >>= 1)
        ps += __shfl_xor(ps, off, 32);
      l[v] = l[v] * alpha[v] + ps;
      m[v] = mnew;
      lds[wave][v + 8 * half][l16]      = p0;   // C-layout -> row-major
      lds[wave][v + 8 * half][16 + l16] = p1;
    }

#pragma unroll
    for (int t = 0; t < 4; ++t)
#pragma unroll
      for (int v = 0; v < 8; ++v) acc[t][v] *= alpha[v];

    // ---- reload P as dense 16x32 A-fragment (two contiguous runs) ----
    // Wave-synchronous: same-wave LDS ops are in order, no barrier needed.
    v4f pc0 = *reinterpret_cast<const v4f*>(&lds[wave][l16][8 * half]);
    v4f pc1 = *reinterpret_cast<const v4f*>(&lds[wave][l16][8 * half + 4]);
    v4f pc2 = *reinterpret_cast<const v4f*>(&lds[wave][l16][16 + 8 * half]);
    v4f pc3 = *reinterpret_cast<const v4f*>(&lds[wave][l16][16 + 8 * half + 4]);
    v16bf pa;
#pragma unroll
    for (int i = 0; i < 4; ++i) {
      pa[i]      = (__bf16)pc0[i];
      pa[4 + i]  = (__bf16)pc1[i];
      pa[8 + i]  = (__bf16)pc2[i];
      pa[12 + i] = (__bf16)pc3[i];
    }

    // ---- acc += P(16x32) * V(32x16) per N-tile; vT is key-contiguous ----
#pragma unroll
    for (int t = 0; t < 4; ++t) {
      v16bf bv = load_bf16x16(vT + vbase + (size_t)(t * 16 + l16) * SEQ +
                              j0 + 8 * half);
      acc[t] = wmma_bf16(pa, bv, acc[t]);
    }
  }

  // z = acc / l  (C/D layout -> [B,T,H] f32)
#pragma unroll
  for (int t = 0; t < 4; ++t)
#pragma unroll
    for (int v = 0; v < 8; ++v) {
      const size_t r = base + (size_t)(qrow0 + v + 8 * half) * HEAD + t * 16 + l16;
      z[r] = acc[t][v] / l[v];
    }
}

// ---------------------------------------------------------------------------
extern "C" void kernel_launch(void* const* d_in, const int* in_sizes, int n_in,
                              void* d_out, int out_size, void* d_ws, size_t ws_size,
                              hipStream_t stream) {
  (void)in_sizes; (void)n_in; (void)out_size; (void)ws_size;
  const float* x  = (const float*)d_in[0];
  const float* Wk = (const float*)d_in[1];
  const float* Wq = (const float*)d_in[2];
  const float* Wv = (const float*)d_in[3];
  float* z = (float*)d_out;
  unsigned short* ws = (unsigned short*)d_ws;   // 3*2MB qkv + 384KB Wt

  // 0) transpose weights to bf16, K-contiguous
  prep_w_kernel<<<(3 * HEAD * EMBED) / 256, 256, 0, stream>>>(Wk, Wq, Wv, ws);
  // 1) projections: 1024 M-tiles / 4 waves per block; y selects k/q/v
  qkv_proj_kernel<<<dim3(256, 3, 1), 128, 0, stream>>>(x, ws);
  // 2) attention: 1024 (batch, query-tile) waves / 8 per block
  flash_attn_kernel<<<128, 256, 0, stream>>>(ws, z);
}
